// HybridTrajNet_37168646980074
// MI455X (gfx1250) — matrix operand order repaired and checked by reading, not compiled, not run on hardware
//
#include <hip/hip_runtime.h>
#include <math.h>

// ---------------------------------------------------------------------------
// HybridTrajNet on gfx1250 (MI455X). GEMMs on v_wmma_f32_16x16x32_bf16,
// B tiles staged block-wide through LDS with gfx1250 async-to-LDS prefetch
// (GLOBAL_LOAD_ASYNC_TO_LDS_B128 + s_wait_asynccnt), sync fallback otherwise.
// ---------------------------------------------------------------------------

typedef __attribute__((ext_vector_type(16))) __bf16 v16bf;
typedef __attribute__((ext_vector_type(8)))  __bf16 bf16x8;
typedef __attribute__((ext_vector_type(8)))  float  f32x8;
typedef __attribute__((ext_vector_type(8)))  float  v8f;

#define ACT_NONE 0
#define ACT_RELU 1
#define ACT_GELU 2

#if defined(__gfx1250__) && __has_builtin(__builtin_amdgcn_global_load_async_to_lds_b128)
#define USE_ASYNC_LDS 1
typedef int i32x4 __attribute__((vector_size(16)));
#define ASYNC_CP_B128(gsrc, ldst) \
    __builtin_amdgcn_global_load_async_to_lds_b128((i32x4*)(gsrc), (i32x4*)(ldst), 0, 0)
#else
#define USE_ASYNC_LDS 0
#endif

__device__ __forceinline__ float apply_act(float v, int act) {
    if (act == ACT_RELU) return fmaxf(v, 0.0f);
    if (act == ACT_GELU) return 0.5f * v * (1.0f + erff(v * 0.70710678118654752f));
    return v;
}

// Stage one 64x32 bf16 B tile (4KB) into LDS. 128 threads, 32B per thread.
// Thread tid covers n_local = tid>>1, k half = (tid&1)*16.
__device__ __forceinline__ void stage_btile(const __bf16* __restrict__ Wb,
                                            __bf16* buf, int nbase, int N, int Kp,
                                            int k0, int tid)
{
    int n = nbase + (tid >> 1);
    if (n >= N) n = N - 1;                 // clamp: junk cols discarded at store
    const __bf16* gsrc = Wb + (size_t)n * Kp + k0 + (tid & 1) * 16;
    __bf16* dst = buf + tid * 16;          // 16 bf16 = 32B per thread
#if USE_ASYNC_LDS
    ASYNC_CP_B128(gsrc,     dst);
    ASYNC_CP_B128(gsrc + 8, dst + 8);
#else
    *(bf16x8*)dst       = *(const bf16x8*)gsrc;
    *(bf16x8*)(dst + 8) = *(const bf16x8*)(gsrc + 8);
#endif
}

__device__ __forceinline__ void wait_stage() {
#if USE_ASYNC_LDS
    asm volatile("s_wait_asynccnt 0x0" ::: "memory");
#endif
}

// ---------------------------------------------------------------------------
// Dense GEMM: C[m, coloff+n] = act( sum_k A[m*lda+k] * Wb[n*Kp+k] + bias[n] )
// A fp32 (M % 64 == 0), Wb bf16 zero-padded (N x Kp, Kp % 32 == 0).
// Block = 4 waves = 4 M-tiles sharing one 64-wide B strip via LDS.
// ---------------------------------------------------------------------------
__global__ void gemm_f32a_wmma(const float* __restrict__ A,
                               const __bf16* __restrict__ Wb,
                               const float* __restrict__ bias,
                               float* __restrict__ C,
                               int M, int N, int Kp,
                               int lda, int ldc, int coloff, int act)
{
    __shared__ __bf16 smem[2 * 64 * 32];   // double-buffered B tile, 8KB
    const int tid  = threadIdx.x;
    const int lane = tid & 31;
    const int wave = tid >> 5;
    const int m16  = lane & 15;
    const int hi   = lane >> 4;
    const int tm   = blockIdx.x * 4 + wave;
    const int nbase = blockIdx.y * 64;

    const float* Arow = A + (size_t)(tm * 16 + m16) * lda;
    v8f acc[4] = {};

    stage_btile(Wb, smem, nbase, N, Kp, 0, tid);
    wait_stage();
    __syncthreads();
    int cur = 0;
    for (int k0 = 0; k0 < Kp; k0 += 32) {
        const bool has_next = (k0 + 32) < Kp;
        if (has_next)
            stage_btile(Wb, smem + (cur ^ 1) * 2048, nbase, N, Kp, k0 + 32, tid);

        const f32x8 alo = *(const f32x8*)(Arow + k0 + hi * 16);
        const f32x8 ahi = *(const f32x8*)(Arow + k0 + hi * 16 + 8);
        v16bf a;
#pragma unroll
        for (int e = 0; e < 8; ++e) { a[e] = (__bf16)alo[e]; a[e + 8] = (__bf16)ahi[e]; }

        // hoist all 4 B fragments (8x ds_load_b128), then issue 4 WMMAs
        const __bf16* bbase = smem + cur * 2048 + m16 * 32 + hi * 16;
        v16bf bfrag[4];
#pragma unroll
        for (int t = 0; t < 4; ++t)
            bfrag[t] = *(const v16bf*)(bbase + t * 16 * 32);
#pragma unroll
        for (int t = 0; t < 4; ++t)
            acc[t] = __builtin_amdgcn_wmma_f32_16x16x32_bf16(
                false, a, false, bfrag[t], (short)0, acc[t], false, false);

        if (has_next) {
            wait_stage();
            __syncthreads();
            cur ^= 1;
        }
    }
#pragma unroll
    for (int t = 0; t < 4; ++t) {
        int n = nbase + t * 16 + m16;
        if (n >= N) continue;
        float bv = bias[n];
#pragma unroll
        for (int e = 0; e < 8; ++e) {
            int row = tm * 16 + hi * 8 + e;
            C[(size_t)row * ldc + coloff + n] = apply_act(acc[t][e] + bv, act);
        }
    }
}

// ---------------------------------------------------------------------------
// bf16 x bf16 GEMM for convs (A = im2col buffer, M x Kp bf16, M % 64 == 0).
// Output row-major M x N (= NHWC activations) with fused bias+ReLU.
// ---------------------------------------------------------------------------
__global__ void gemm_bf16a_wmma(const __bf16* __restrict__ Ab,
                                const __bf16* __restrict__ Wb,
                                const float* __restrict__ bias,
                                float* __restrict__ C,
                                int M, int N, int Kp)
{
    __shared__ __bf16 smem[2 * 64 * 32];
    const int tid  = threadIdx.x;
    const int lane = tid & 31;
    const int wave = tid >> 5;
    const int m16  = lane & 15;
    const int hi   = lane >> 4;
    const int tm   = blockIdx.x * 4 + wave;
    const int nbase = blockIdx.y * 64;

    const __bf16* Arow = Ab + (size_t)(tm * 16 + m16) * Kp;
    v8f acc[4] = {};

    stage_btile(Wb, smem, nbase, N, Kp, 0, tid);
    wait_stage();
    __syncthreads();
    int cur = 0;
    for (int k0 = 0; k0 < Kp; k0 += 32) {
        const bool has_next = (k0 + 32) < Kp;
        if (has_next)
            stage_btile(Wb, smem + (cur ^ 1) * 2048, nbase, N, Kp, k0 + 32, tid);

        const bf16x8 alo = *(const bf16x8*)(Arow + k0 + hi * 16);
        const bf16x8 ahi = *(const bf16x8*)(Arow + k0 + hi * 16 + 8);
        v16bf a;
#pragma unroll
        for (int e = 0; e < 8; ++e) { a[e] = alo[e]; a[e + 8] = ahi[e]; }

        const __bf16* bbase = smem + cur * 2048 + m16 * 32 + hi * 16;
        v16bf bfrag[4];
#pragma unroll
        for (int t = 0; t < 4; ++t)
            bfrag[t] = *(const v16bf*)(bbase + t * 16 * 32);
#pragma unroll
        for (int t = 0; t < 4; ++t)
            acc[t] = __builtin_amdgcn_wmma_f32_16x16x32_bf16(
                false, a, false, bfrag[t], (short)0, acc[t], false, false);

        if (has_next) {
            wait_stage();
            __syncthreads();
            cur ^= 1;
        }
    }
#pragma unroll
    for (int t = 0; t < 4; ++t) {
        int n = nbase + t * 16 + m16;
        if (n >= N) continue;
        float bv = bias[n];
#pragma unroll
        for (int e = 0; e < 8; ++e) {
            int row = tm * 16 + hi * 8 + e;
            C[(size_t)row * (size_t)N + n] = fmaxf(acc[t][e] + bv, 0.0f);
        }
    }
}

// ---------------------------------------------------------------------------
// Weight staging: fp32 -> zero-padded bf16
// ---------------------------------------------------------------------------
__global__ void cvtw_dense_kernel(const float* __restrict__ W, __bf16* __restrict__ Wb,
                                  int N, int K, int Kp)
{
    int idx = blockIdx.x * blockDim.x + threadIdx.x;
    if (idx >= N * Kp) return;
    int n = idx / Kp, k = idx % Kp;
    Wb[idx] = (k < K) ? (__bf16)W[(size_t)n * K + k] : (__bf16)0.0f;
}

// Conv weight (Co,Ci,3,3) -> bf16 (Co x Kp), K reordered as (r*3+s)*Ci + ci
__global__ void cvtw_conv_kernel(const float* __restrict__ W, __bf16* __restrict__ Wb,
                                 int Co, int Ci, int Kp)
{
    int idx = blockIdx.x * blockDim.x + threadIdx.x;
    if (idx >= Co * Kp) return;
    int n = idx / Kp, k = idx % Kp;
    __bf16 v = (__bf16)0.0f;
    if (k < Ci * 9) {
        int rs = k / Ci, ci = k % Ci;
        v = (__bf16)W[((size_t)n * Ci + ci) * 9 + rs];
    }
    Wb[idx] = v;
}

// ---------------------------------------------------------------------------
// im2col (NHWC, 3x3 stride 2 pad 1) -> bf16 column buffer (M x Kp),
// K ordered (r*3+s)*Ci + ci so ci runs are contiguous in memory.
// ---------------------------------------------------------------------------
__global__ void im2col_kernel(const float* __restrict__ X, __bf16* __restrict__ col,
                              int Ci, int H, int Wi, int Ho, int Wo, int Kp,
                              long long total)
{
    long long idx = (long long)blockIdx.x * blockDim.x + threadIdx.x;
    if (idx >= total) return;
    int k = (int)(idx % Kp);
    long long m = idx / Kp;
    int ox = (int)(m % Wo);
    long long t2 = m / Wo;
    int oy = (int)(t2 % Ho);
    int b  = (int)(t2 / Ho);
    float v = 0.0f;
    if (k < Ci * 9) {
        int rs = k / Ci, ci = k % Ci;
        int r = rs / 3, s = rs % 3;
        int iy = oy * 2 - 1 + r, ix = ox * 2 - 1 + s;
        if (iy >= 0 && iy < H && ix >= 0 && ix < Wi)
            v = X[(((size_t)b * H + iy) * Wi + ix) * Ci + ci];
    }
    col[idx] = (__bf16)v;
}

// ---------------------------------------------------------------------------
// Rasterize: hist (512,50,50,6) -> img NHWC (512,128,128,3)
// ---------------------------------------------------------------------------
__global__ void rasterize_kernel(const float* __restrict__ hist,
                                 float* __restrict__ img)
{
    int idx = blockIdx.x * blockDim.x + threadIdx.x;
    if (idx >= 512 * 50 * 50) return;
    int b = idx / 2500;
    int t = idx % 50;
    const float* h = hist + (size_t)idx * 6;
    float x = h[0], y = h[1], vx = h[2], vy = h[3];
    int xp = (int)rintf(x * 2.0f + 63.5f);
    int yp = (int)rintf(y * 2.0f + 63.5f);
    if (xp < 0 || xp >= 128 || yp < 0 || yp >= 128) return;
    float tval = (float)t * (1.0f / 49.0f);
    float ch   = cosf(atan2f(vy, vx));
    float* cell = img + ((size_t)b * 128 * 128 + yp * 128 + xp) * 3;
    atomicAdd(&cell[0], 1.0f);
    cell[1] = tval;
    cell[2] = ch;
}

__global__ void clamp01_kernel(float* __restrict__ p, int n) {
    int i = blockIdx.x * blockDim.x + threadIdx.x;
    if (i < n) p[i] = fminf(fmaxf(p[i], 0.0f), 1.0f);
}

// Max over spatial: h4 NHWC (512, 64 positions, 128 ch) -> (512,128)
__global__ void maxpool_kernel(const float* __restrict__ X, float* __restrict__ Y) {
    int idx = blockIdx.x * blockDim.x + threadIdx.x;
    if (idx >= 512 * 128) return;
    int b = idx / 128, c = idx % 128;
    const float* p = X + (size_t)b * 64 * 128 + c;
    float m = p[0];
#pragma unroll 8
    for (int i = 1; i < 64; ++i) m = fmaxf(m, p[(size_t)i * 128]);
    Y[idx] = m;
}

// Node features -> (512,50,32) zero-padded rows (buffer pre-zeroed)
__global__ void node_kernel(const float* __restrict__ hist,
                            const float* __restrict__ scale,
                            float* __restrict__ node)
{
    int idx = blockIdx.x * blockDim.x + threadIdx.x;
    if (idx >= 512 * 50) return;
    int b = idx / 50, a = idx % 50;
    const float* last = hist + (((size_t)(b * 50 + a)) * 50 + 49) * 6;
    const float* ego  = hist + (((size_t)(b * 50 + 0)) * 50 + 49) * 6;
    float sc = scale[b];
    float rx = last[0] - ego[0], ry = last[1] - ego[1];
    float r   = sqrtf(rx * sc * rx * sc + ry * sc * ry * sc);
    float phi = atan2f(ry, rx);
    float* o = node + (size_t)idx * 32;
    o[0] = rx;            o[1] = ry;
    o[2] = last[2] * sc;  o[3] = last[3] * sc;
    o[4] = last[4] * sc;  o[5] = last[5] * sc;
    o[6] = r;             o[7] = cosf(phi);
    o[8] = sinf(phi);     o[9] = (a == 0) ? 1.0f : 0.0f;
}

// Attention scores (512,4,50,50), dh=32
__global__ void attn_scores_kernel(const float* __restrict__ qkv,
                                   float* __restrict__ scores)
{
    int idx = blockIdx.x * blockDim.x + threadIdx.x;
    if (idx >= 512 * 4 * 50 * 50) return;
    int kk = idx % 50; int tmp = idx / 50;
    int q  = tmp % 50; tmp /= 50;
    int h  = tmp % 4;  int b = tmp / 4;
    const float* qp = qkv + ((size_t)(b * 50 + q)) * 384 + h * 32;
    const float* kp = qkv + ((size_t)(b * 50 + kk)) * 384 + 128 + h * 32;
    float s = 0.0f;
#pragma unroll
    for (int d = 0; d < 32; ++d) s += qp[d] * kp[d];
    scores[idx] = s * 0.176776695296636881f;
}

__global__ void softmax50_kernel(float* __restrict__ scores) {
    int row = blockIdx.x;
    float* p = scores + (size_t)row * 50;
    int lane = threadIdx.x;
    float v0 = (lane < 50)      ? p[lane]      : -INFINITY;
    float v1 = (lane + 32 < 50) ? p[lane + 32] : -INFINITY;
    float m = fmaxf(v0, v1);
    for (int o = 16; o > 0; o >>= 1) m = fmaxf(m, __shfl_xor(m, o, 32));
    float e0 = (lane < 50)      ? expf(v0 - m) : 0.0f;
    float e1 = (lane + 32 < 50) ? expf(v1 - m) : 0.0f;
    float s = e0 + e1;
    for (int o = 16; o > 0; o >>= 1) s += __shfl_xor(s, o, 32);
    float inv = 1.0f / s;
    if (lane < 50)      p[lane]      = e0 * inv;
    if (lane + 32 < 50) p[lane + 32] = e1 * inv;
}

__global__ void attn_combine_kernel(const float* __restrict__ qkv,
                                    const float* __restrict__ scores,
                                    float* __restrict__ o)
{
    int idx = blockIdx.x * blockDim.x + threadIdx.x;
    if (idx >= 512 * 50 * 128) return;
    int c = idx % 128; int row = idx / 128;
    int q = row % 50,  b = row / 50;
    int h = c / 32;
    const float* att = scores + ((size_t)((b * 4 + h) * 50 + q)) * 50;
    const float* vb  = qkv + (size_t)(b * 50) * 384 + 256 + c;
    float s = 0.0f;
#pragma unroll 10
    for (int k = 0; k < 50; ++k) s += att[k] * vb[(size_t)k * 384];
    o[idx] = s;
}

// x = LN(x + y) * g + b; rows of 128; one wave per row
__global__ void add_ln_kernel(float* __restrict__ x, const float* __restrict__ y,
                              const float* __restrict__ g, const float* __restrict__ bta)
{
    int row  = blockIdx.x;
    int lane = threadIdx.x;
    float* xp = x + (size_t)row * 128;
    const float* yp = y + (size_t)row * 128;
    float v[4];
    float sum = 0.0f;
#pragma unroll
    for (int i = 0; i < 4; ++i) { v[i] = xp[lane + i * 32] + yp[lane + i * 32]; sum += v[i]; }
    for (int o = 16; o > 0; o >>= 1) sum += __shfl_xor(sum, o, 32);
    float mean = sum * (1.0f / 128.0f);
    float var = 0.0f;
#pragma unroll
    for (int i = 0; i < 4; ++i) { float d = v[i] - mean; var += d * d; }
    for (int o = 16; o > 0; o >>= 1) var += __shfl_xor(var, o, 32);
    float inv = rsqrtf(var * (1.0f / 128.0f) + 1e-5f);
#pragma unroll
    for (int i = 0; i < 4; ++i) {
        int c = lane + i * 32;
        xp[c] = (v[i] - mean) * inv * g[c] + bta[c];
    }
}

// hist[:,0] -> (512,50,32) zero-padded rows (buffer pre-zeroed)
__global__ void egoseq_kernel(const float* __restrict__ hist, float* __restrict__ seq) {
    int idx = blockIdx.x * blockDim.x + threadIdx.x;
    if (idx >= 512 * 50 * 6) return;
    int c = idx % 6; int row = idx / 6;
    int t = row % 50, b = row / 50;
    seq[(size_t)row * 32 + c] = hist[(((size_t)b * 50 + 0) * 50 + t) * 6 + c];
}

__global__ void lstm_cell_kernel(const float* __restrict__ xproj,
                                 const float* __restrict__ hproj,
                                 float* __restrict__ h, float* __restrict__ c,
                                 int t)
{
    int idx = blockIdx.x * blockDim.x + threadIdx.x;
    if (idx >= 512 * 256) return;
    int b = idx / 256, j = idx % 256;
    const float* xg = xproj + ((size_t)b * 50 + t) * 1024;
    const float* hg = hproj + (size_t)b * 1024;
    float gi = xg[j]       + hg[j];
    float gf = xg[256 + j] + hg[256 + j];
    float gg = xg[512 + j] + hg[512 + j];
    float go = xg[768 + j] + hg[768 + j];
    float si = 1.0f / (1.0f + expf(-gi));
    float sf = 1.0f / (1.0f + expf(-gf));
    float so = 1.0f / (1.0f + expf(-go));
    float cc = sf * c[idx] + si * tanhf(gg);
    c[idx] = cc;
    h[idx] = so * tanhf(cc);
}

__global__ void copy_ego_kernel(const float* __restrict__ h, float* __restrict__ fused) {
    int idx = blockIdx.x * blockDim.x + threadIdx.x;
    if (idx >= 512 * 256) return;
    int b = idx / 256, j = idx % 256;
    fused[(size_t)b * 768 + 512 + j] = h[idx];
}

// ---------------------------------------------------------------------------
extern "C" void kernel_launch(void* const* d_in, const int* in_sizes, int n_in,
                              void* d_out, int out_size, void* d_ws, size_t ws_size,
                              hipStream_t stream)
{
    const float* hist  = (const float*)d_in[0];
    const float* scale = (const float*)d_in[1];
    const float* conv_w[4], *conv_b[4];
    for (int i = 0; i < 4; ++i) { conv_w[i] = (const float*)d_in[2 + i]; conv_b[i] = (const float*)d_in[6 + i]; }
    const float* road_w  = (const float*)d_in[10];
    const float* road_b  = (const float*)d_in[11];
    const float* infc_w  = (const float*)d_in[12];
    const float* infc_b  = (const float*)d_in[13];
    struct Enc { const float *w_qkv,*b_qkv,*w_o,*b_o,*ln1_g,*ln1_b,*w_ff1,*b_ff1,*w_ff2,*b_ff2,*ln2_g,*ln2_b; };
    Enc enc[3];
    for (int l = 0; l < 3; ++l) {
        int k = 14 + 12 * l;
        enc[l].w_qkv = (const float*)d_in[k+0];  enc[l].b_qkv = (const float*)d_in[k+1];
        enc[l].w_o   = (const float*)d_in[k+2];  enc[l].b_o   = (const float*)d_in[k+3];
        enc[l].ln1_g = (const float*)d_in[k+4];  enc[l].ln1_b = (const float*)d_in[k+5];
        enc[l].w_ff1 = (const float*)d_in[k+6];  enc[l].b_ff1 = (const float*)d_in[k+7];
        enc[l].w_ff2 = (const float*)d_in[k+8];  enc[l].b_ff2 = (const float*)d_in[k+9];
        enc[l].ln2_g = (const float*)d_in[k+10]; enc[l].ln2_b = (const float*)d_in[k+11];
    }
    const float* soc_w  = (const float*)d_in[50];
    const float* soc_b  = (const float*)d_in[51];
    const float* wih    = (const float*)d_in[52];
    const float* whh    = (const float*)d_in[53];
    const float* bih    = (const float*)d_in[54];
    const float* bhh    = (const float*)d_in[55];
    const float* fu1_w  = (const float*)d_in[56];
    const float* fu1_b  = (const float*)d_in[57];
    const float* fu2_w  = (const float*)d_in[58];
    const float* fu2_b  = (const float*)d_in[59];
    const float* dec_w  = (const float*)d_in[60];
    const float* dec_b  = (const float*)d_in[61];
    float* out = (float*)d_out;

    // ---- workspace carve-up (256B aligned) ----
    char* ws = (char*)d_ws;
    size_t off = 0;
    auto alloc = [&](size_t bytes) -> void* {
        void* p = ws + off;
        off = (off + bytes + 255) & ~(size_t)255;
        return p;
    };
    float* img    = (float*)alloc((size_t)512*128*128*3*4);     // NHWC
    __bf16* col   = (__bf16*)alloc((size_t)512*32*32*288*2);    // reusable im2col (max: conv2)
    float* h1     = (float*)alloc((size_t)512*64*64*32*4);      // NHWC
    float* h2     = (float*)alloc((size_t)512*32*32*64*4);
    float* h3     = (float*)alloc((size_t)512*16*16*128*4);
    float* h4     = (float*)alloc((size_t)512*8*8*128*4);
    float* pool   = (float*)alloc((size_t)512*128*4);
    float* node   = (float*)alloc((size_t)512*50*32*4);
    float* x      = (float*)alloc((size_t)512*50*128*4);
    float* qkv    = (float*)alloc((size_t)512*50*384*4);
    float* scores = (float*)alloc((size_t)512*4*50*50*4);
    float* obuf   = (float*)alloc((size_t)512*50*128*4);
    float* proj   = (float*)alloc((size_t)512*50*128*4);
    float* ff     = (float*)alloc((size_t)512*50*256*4);
    float* ffo    = (float*)alloc((size_t)512*50*128*4);
    float* egoseq = (float*)alloc((size_t)512*50*32*4);
    float* xproj  = (float*)alloc((size_t)512*50*1024*4);
    float* hstate = (float*)alloc((size_t)512*256*4);
    float* cstate = (float*)alloc((size_t)512*256*4);
    float* hproj  = (float*)alloc((size_t)512*1024*4);
    float* fused  = (float*)alloc((size_t)512*768*4);
    float* f1     = (float*)alloc((size_t)512*512*4);
    float* f2     = (float*)alloc((size_t)512*512*4);
    // bf16 weight staging
    __bf16* cw_b[4];
    int convKp[4] = { 32, 288, 576, 1152 };
    int convCo[4] = { 32, 64, 128, 128 };
    int convCi[4] = { 3, 32, 64, 128 };
    for (int i = 0; i < 4; ++i) cw_b[i] = (__bf16*)alloc((size_t)convCo[i]*convKp[i]*2);
    __bf16* road_wb = (__bf16*)alloc((size_t)256*128*2);
    __bf16* infc_wb = (__bf16*)alloc((size_t)128*32*2);
    __bf16* qkv_wb[3], *wo_wb[3], *ff1_wb[3], *ff2_wb[3];
    for (int l = 0; l < 3; ++l) {
        qkv_wb[l] = (__bf16*)alloc((size_t)384*128*2);
        wo_wb[l]  = (__bf16*)alloc((size_t)128*128*2);
        ff1_wb[l] = (__bf16*)alloc((size_t)256*128*2);
        ff2_wb[l] = (__bf16*)alloc((size_t)128*256*2);
    }
    __bf16* soc_wb = (__bf16*)alloc((size_t)256*128*2);
    __bf16* wih_b  = (__bf16*)alloc((size_t)1024*32*2);
    __bf16* whh_b  = (__bf16*)alloc((size_t)1024*256*2);
    __bf16* fu1_wb = (__bf16*)alloc((size_t)512*768*2);
    __bf16* fu2_wb = (__bf16*)alloc((size_t)512*512*2);
    __bf16* dec_wb = (__bf16*)alloc((size_t)120*512*2);

    auto cvtw = [&](const float* W, __bf16* Wb, int N, int K, int Kp) {
        cvtw_dense_kernel<<<(N * Kp + 255) / 256, 256, 0, stream>>>(W, Wb, N, K, Kp);
    };
    auto gemm = [&](const float* A, const __bf16* Wb, const float* bias, float* C,
                    int M, int N, int Kp, int lda, int ldc, int coloff, int act) {
        dim3 grid(M / 64, (N + 63) / 64);   // all M are multiples of 64
        gemm_f32a_wmma<<<grid, dim3(128), 0, stream>>>(A, Wb, bias, C, M, N, Kp, lda, ldc, coloff, act);
    };

    // ---- 0. weight staging ----
    for (int i = 0; i < 4; ++i)
        cvtw_conv_kernel<<<(convCo[i]*convKp[i] + 255) / 256, 256, 0, stream>>>(
            conv_w[i], cw_b[i], convCo[i], convCi[i], convKp[i]);
    cvtw(road_w, road_wb, 256, 128, 128);
    cvtw(infc_w, infc_wb, 128, 10, 32);
    for (int l = 0; l < 3; ++l) {
        cvtw(enc[l].w_qkv, qkv_wb[l], 384, 128, 128);
        cvtw(enc[l].w_o,   wo_wb[l],  128, 128, 128);
        cvtw(enc[l].w_ff1, ff1_wb[l], 256, 128, 128);
        cvtw(enc[l].w_ff2, ff2_wb[l], 128, 256, 256);
    }
    cvtw(soc_w, soc_wb, 256, 128, 128);
    cvtw(wih,   wih_b,  1024, 6, 32);
    cvtw(whh,   whh_b,  1024, 256, 256);
    cvtw(fu1_w, fu1_wb, 512, 768, 768);
    cvtw(fu2_w, fu2_wb, 512, 512, 512);
    cvtw(dec_w, dec_wb, 120, 512, 512);

    // ---- 1. rasterize ----
    (void)hipMemsetAsync(img, 0, (size_t)512*128*128*3*4, stream);
    rasterize_kernel<<<(512*50*50 + 255) / 256, 256, 0, stream>>>(hist, img);
    clamp01_kernel<<<(512*3*128*128 + 255) / 256, 256, 0, stream>>>(img, 512*3*128*128);

    // ---- 2. CNN (im2col + bf16 WMMA GEMM), all NHWC ----
    {
        const float* src[4] = { img, h1, h2, h3 };
        float*       dst[4] = { h1, h2, h3, h4 };
        int H[4] = { 128, 64, 32, 16 };
        for (int i = 0; i < 4; ++i) {
            int Ho = H[i] / 2, Wo = H[i] / 2;
            long long total = (long long)512 * Ho * Wo * convKp[i];
            im2col_kernel<<<(unsigned)((total + 255) / 256), 256, 0, stream>>>(
                src[i], col, convCi[i], H[i], H[i], Ho, Wo, convKp[i], total);
            int M = 512 * Ho * Wo;
            dim3 grid(M / 64, (convCo[i] + 63) / 64);
            gemm_bf16a_wmma<<<grid, dim3(128), 0, stream>>>(
                col, cw_b[i], conv_b[i], dst[i], M, convCo[i], convKp[i]);
        }
    }
    maxpool_kernel<<<(512*128 + 255) / 256, 256, 0, stream>>>(h4, pool);
    gemm(pool, road_wb, road_b, fused, 512, 256, 128, 128, 768, 0, ACT_NONE);

    // ---- 3. social transformer ----
    (void)hipMemsetAsync(node, 0, (size_t)512*50*32*4, stream);
    node_kernel<<<(512*50 + 255) / 256, 256, 0, stream>>>(hist, scale, node);
    gemm(node, infc_wb, infc_b, x, 25600, 128, 32, 32, 128, 0, ACT_NONE);
    for (int l = 0; l < 3; ++l) {
        gemm(x, qkv_wb[l], enc[l].b_qkv, qkv, 25600, 384, 128, 128, 384, 0, ACT_NONE);
        attn_scores_kernel<<<(512*4*50*50 + 255) / 256, 256, 0, stream>>>(qkv, scores);
        softmax50_kernel<<<512*4*50, 32, 0, stream>>>(scores);
        attn_combine_kernel<<<(512*50*128 + 255) / 256, 256, 0, stream>>>(qkv, scores, obuf);
        gemm(obuf, wo_wb[l], enc[l].b_o, proj, 25600, 128, 128, 128, 128, 0, ACT_NONE);
        add_ln_kernel<<<25600, 32, 0, stream>>>(x, proj, enc[l].ln1_g, enc[l].ln1_b);
        gemm(x, ff1_wb[l], enc[l].b_ff1, ff, 25600, 256, 128, 128, 256, 0, ACT_GELU);
        gemm(ff, ff2_wb[l], enc[l].b_ff2, ffo, 25600, 128, 256, 256, 128, 0, ACT_NONE);
        add_ln_kernel<<<25600, 32, 0, stream>>>(x, ffo, enc[l].ln2_g, enc[l].ln2_b);
    }
    // social_f = x[:,0,:] @ soc_w.T -> fused[:,256:512]  (A row stride = 50*128)
    gemm(x, soc_wb, soc_b, fused, 512, 256, 128, 50*128, 768, 256, ACT_NONE);

    // ---- 4. ego LSTM ----
    (void)hipMemsetAsync(egoseq, 0, (size_t)512*50*32*4, stream);
    egoseq_kernel<<<(512*50*6 + 255) / 256, 256, 0, stream>>>(hist, egoseq);
    gemm(egoseq, wih_b, bih, xproj, 25600, 1024, 32, 32, 1024, 0, ACT_NONE);
    (void)hipMemsetAsync(hstate, 0, (size_t)512*256*4, stream);
    (void)hipMemsetAsync(cstate, 0, (size_t)512*256*4, stream);
    for (int t = 0; t < 50; ++t) {
        gemm(hstate, whh_b, bhh, hproj, 512, 1024, 256, 256, 1024, 0, ACT_NONE);
        lstm_cell_kernel<<<(512*256 + 255) / 256, 256, 0, stream>>>(xproj, hproj, hstate, cstate, t);
    }
    copy_ego_kernel<<<(512*256 + 255) / 256, 256, 0, stream>>>(hstate, fused);

    // ---- 5. fusion + decode ----
    gemm(fused, fu1_wb, fu1_b, f1, 512, 512, 768, 768, 512, 0, ACT_RELU);
    gemm(f1,    fu2_wb, fu2_b, f2, 512, 512, 512, 512, 512, 0, ACT_RELU);
    gemm(f2,    dec_wb, dec_b, out, 512, 120, 512, 512, 120, 0, ACT_NONE);
}
